// EZGNN_64510408786461
// MI455X (gfx1250) — matrix operand order repaired and checked by reading, not compiled, not run on hardware
//
#include <hip/hip_runtime.h>
#include <hip/hip_bf16.h>
#include <math.h>

#define NN     50000
#define NE     800000
#define NEL    (NE + NN)      /* 850000, multiple of 16 */
#define KNODE  40
#define KEDGE  68
#define HD     64
#define NL     4
#define NG     64
#define LN_EPS 1e-5f

typedef __attribute__((ext_vector_type(16))) _Float16 v16h;
typedef __attribute__((ext_vector_type(8)))  float    v8f;

// ---------------- WMMA fragment helpers (v_wmma_f32_16x16x32_f16) -----------
// A 16x32 f16: lane r (0-15) = row r ; lanes 0-15 hold K {k0..k0+7},{k0+16..k0+23},
// lanes 16-31 hold K {k0+8..k0+15},{k0+24..k0+31}; rowptr = per-lane row base.
__device__ __forceinline__ v16h frag_a(const _Float16* rowptr, int k0) {
  const int h = (threadIdx.x & 31) >> 4;
  const _Float16* p = rowptr + k0 + h * 8;
  v16h a;
#pragma unroll
  for (int i = 0; i < 8; ++i) a[i] = p[i];
#pragma unroll
  for (int i = 0; i < 8; ++i) a[8 + i] = p[16 + i];
  return a;
}

// B 32x16 f16 from transposed weights Bt[n][k] (row length Kld):
// lane = column n (mod 16); lanes 0-15 K {k0..k0+7},{k0+16..k0+23}, lanes 16-31 K {k0+8..k0+15},{k0+24..k0+31}.
__device__ __forceinline__ v16h frag_b(const _Float16* Bt, int Kld, int k0, int n0) {
  const int lane = threadIdx.x & 31;
  const int n = lane & 15, h = lane >> 4;
  const _Float16* p = Bt + (size_t)(n0 + n) * Kld + k0 + h * 8;
  v16h b;
#pragma unroll
  for (int i = 0; i < 8; ++i) { b[i] = p[i]; b[8 + i] = p[16 + i]; }
  return b;
}

// C 16x16 f32: VGPR g -> row (h*8+g), col (lane&15)
__device__ __forceinline__ v8f frag_bias(const float* bias, int n0) {
  const float bv = bias[n0 + (threadIdx.x & 15)];
  v8f c;
#pragma unroll
  for (int g = 0; g < 8; ++g) c[g] = bv;
  return c;
}

#define WMMA_F16(A, B, C) \
  __builtin_amdgcn_wmma_f32_16x16x32_f16(false, (A), false, (B), (short)0, (C), false, false)

// ---------------- weight convert: W[K][N] f32 -> Bt[N][Kpad] f16 (zero pad) --
__global__ void k_conv_bt(const float* __restrict__ W, _Float16* __restrict__ Bt,
                          int K, int N, int Kpad) {
  int idx = blockIdx.x * blockDim.x + threadIdx.x;
  if (idx >= N * Kpad) return;
  int n = idx / Kpad, k = idx - n * Kpad;
  Bt[idx] = (k < K) ? (_Float16)W[(size_t)k * N + n] : (_Float16)0.f;
}

// ---------------- build src/dst with self-loops; zero ea16 tail -------------
__global__ void k_build_edges(const int* __restrict__ ei, int* __restrict__ src,
                              int* __restrict__ dst) {
  int i = blockIdx.x * blockDim.x + threadIdx.x;
  if (i >= NEL) return;
  if (i < NE) { src[i] = ei[i]; dst[i] = ei[NE + i]; }
  else        { src[i] = i - NE; dst[i] = i - NE; }
}
__global__ void k_zero_ea_tail(_Float16* __restrict__ ea16) {
  int i = blockIdx.x * blockDim.x + threadIdx.x;
  if (i < NN * HD) ea16[(size_t)NE * HD + i] = (_Float16)0.f;
}

// ---------------- input projection: X[nRows][Kin] @ Bt -> 64 cols ------------
// per-wave LDS staging with zero-padded K (Kpad = 64 or 96)
__global__ void __launch_bounds__(256)
k_in_proj(const float* __restrict__ X, const _Float16* __restrict__ wt,
          const float* __restrict__ bias, float* __restrict__ outF,
          _Float16* __restrict__ outH, int Kin, int Kpad, int nRows) {
  __shared__ _Float16 s_a[8][16 * 96];
  const int wave = threadIdx.x >> 5, lane = threadIdx.x & 31;
  const int m = lane & 15, h = lane >> 4;
  const int r0 = (blockIdx.x * 8 + wave) * 16;
  if (r0 >= nRows) return;
  _Float16* sa = s_a[wave];
  for (int t = lane; t < 16 * Kpad; t += 32) {
    int r = t / Kpad, c = t - r * Kpad;
    sa[t] = (c < Kin) ? (_Float16)X[(size_t)(r0 + r) * Kin + c] : (_Float16)0.f;
  }
  asm volatile("s_wait_dscnt 0" ::: "memory");
  const _Float16* pa = sa + m * Kpad;
  const int kch = Kpad >> 5;
#pragma unroll
  for (int nt = 0; nt < 4; ++nt) {
    v8f c = frag_bias(bias, nt * 16);
    for (int kc = 0; kc < kch; ++kc)
      c = WMMA_F16(frag_a(pa, kc * 32), frag_b(wt, Kpad, kc * 32, nt * 16), c);
    const int col = nt * 16 + (lane & 15);
#pragma unroll
    for (int g = 0; g < 8; ++g) {
      size_t o = (size_t)(r0 + h * 8 + g) * HD + col;
      if (outF) outF[o] = c[g];
      if (outH) outH[o] = (_Float16)c[g];
    }
  }
}

// ---------------- node linear: x16[NN][64] @ Bt[64][64] + b -> f16 ----------
__global__ void __launch_bounds__(256)
k_node_lin(const _Float16* __restrict__ x16, const _Float16* __restrict__ wt,
           const float* __restrict__ bias, _Float16* __restrict__ out16, int nRows) {
  __shared__ _Float16 s_w[HD * HD];
  for (int t = threadIdx.x; t < HD * HD; t += 256) s_w[t] = wt[t];
  __syncthreads();
  const int wave = threadIdx.x >> 5, lane = threadIdx.x & 31;
  const int m = lane & 15, h = lane >> 4;
  const int r0 = (blockIdx.x * 8 + wave) * 16;
  if (r0 >= nRows) return;
  const _Float16* pa = x16 + (size_t)(r0 + m) * HD;
#pragma unroll
  for (int nt = 0; nt < 4; ++nt) {
    v8f c = frag_bias(bias, nt * 16);
#pragma unroll
    for (int kc = 0; kc < 2; ++kc)
      c = WMMA_F16(frag_a(pa, kc * 32), frag_b(s_w, HD, kc * 32, nt * 16), c);
    const int col = nt * 16 + (lane & 15);
#pragma unroll
    for (int g = 0; g < 8; ++g)
      out16[(size_t)(r0 + h * 8 + g) * HD + col] = (_Float16)c[g];
  }
}

// ---------------- edge MLP: gather -> GEMM(192->64) relu -> GEMM(64->64) ----
// -> atomic scatter-add into aggr[dst]
__global__ void __launch_bounds__(256)
k_edge_mlp(const _Float16* __restrict__ xs16, const _Float16* __restrict__ xd16,
           const _Float16* __restrict__ ea16, const int* __restrict__ src,
           const int* __restrict__ dst, const _Float16* __restrict__ w1t,
           const float* __restrict__ b1, const _Float16* __restrict__ w2t,
           const float* __restrict__ b2, float* __restrict__ aggr, int nE) {
  __shared__ _Float16 s_w1[HD * 192];
  __shared__ _Float16 s_w2[HD * HD];
  __shared__ _Float16 s_hid[8][16 * HD];
  for (int t = threadIdx.x; t < HD * 192; t += 256) s_w1[t] = w1t[t];
  for (int t = threadIdx.x; t < HD * HD;  t += 256) s_w2[t] = w2t[t];
  __syncthreads();
  const int wave = threadIdx.x >> 5, lane = threadIdx.x & 31;
  const int m = lane & 15, h = lane >> 4;
  const int e0 = (blockIdx.x * 8 + wave) * 16;
  if (e0 >= nE) return;

  const _Float16* pa_s = xs16 + (size_t)src[e0 + m] * HD;
  const _Float16* pa_d = xd16 + (size_t)dst[e0 + m] * HD;
  const _Float16* pa_e = ea16 + (size_t)(e0 + m) * HD;
  int drow[8];
#pragma unroll
  for (int g = 0; g < 8; ++g) drow[g] = dst[e0 + h * 8 + g];

  _Float16* hid = s_hid[wave];
  // hidden = relu([xs|xd|ea] @ W1 + b1), K = 192
#pragma unroll
  for (int nt = 0; nt < 4; ++nt) {
    v8f c = frag_bias(b1, nt * 16);
#pragma unroll
    for (int kc = 0; kc < 2; ++kc) {
      c = WMMA_F16(frag_a(pa_s, kc * 32), frag_b(s_w1, 192,       kc * 32, nt * 16), c);
      c = WMMA_F16(frag_a(pa_d, kc * 32), frag_b(s_w1, 192,  64 + kc * 32, nt * 16), c);
      c = WMMA_F16(frag_a(pa_e, kc * 32), frag_b(s_w1, 192, 128 + kc * 32, nt * 16), c);
    }
    const int col = nt * 16 + (lane & 15);
#pragma unroll
    for (int g = 0; g < 8; ++g)
      hid[(h * 8 + g) * HD + col] = (_Float16)fmaxf(c[g], 0.f);
  }
  asm volatile("s_wait_dscnt 0" ::: "memory");
  const _Float16* pa_h = hid + m * HD;
  // out = hidden @ W2 + b2, scatter-add
#pragma unroll
  for (int nt = 0; nt < 4; ++nt) {
    v8f c = frag_bias(b2, nt * 16);
#pragma unroll
    for (int kc = 0; kc < 2; ++kc)
      c = WMMA_F16(frag_a(pa_h, kc * 32), frag_b(s_w2, HD, kc * 32, nt * 16), c);
    const int col = nt * 16 + (lane & 15);
#pragma unroll
    for (int g = 0; g < 8; ++g)
      atomicAdd(&aggr[(size_t)drow[g] * HD + col], c[g]);
  }
}

// ---------------- elementwise helpers ---------------------------------------
__global__ void k_f32_to_f16(const float* __restrict__ a, _Float16* __restrict__ b, int n) {
  int i = blockIdx.x * blockDim.x + threadIdx.x;
  if (i < n) b[i] = (_Float16)a[i];
}
__global__ void k_x_update(float* __restrict__ x, const float* __restrict__ aggr, int n) {
  int i = blockIdx.x * blockDim.x + threadIdx.x;
  if (i < n) x[i] = fmaxf(x[i] + aggr[i], 0.f);
}

// ---------------- pooling + heads -------------------------------------------
__global__ void k_pool(const float* __restrict__ x, const int* __restrict__ batch,
                       float* __restrict__ sums, float* __restrict__ cnt) {
  int i = blockIdx.x * blockDim.x + threadIdx.x;
  if (i >= NN * HD) return;
  int node = i >> 6, j = i & 63;
  int b = batch[node];
  atomicAdd(&sums[b * HD + j], x[i]);
  if (j == 0) atomicAdd(&cnt[b], 1.f);
}

__global__ void __launch_bounds__(256)
k_node_head(const float* __restrict__ x, const float* __restrict__ ln_g,
            const float* __restrict__ ln_b, const float* __restrict__ w1,
            const float* __restrict__ b1, const float* __restrict__ w2,
            const float* __restrict__ b2, float* __restrict__ out) {
  __shared__ float sw1[HD * 32], sb1[32], sw2[32], sg[HD], sb[HD];
  for (int t = threadIdx.x; t < HD * 32; t += 256) sw1[t] = w1[t];
  if (threadIdx.x < 32) { sb1[threadIdx.x] = b1[threadIdx.x]; sw2[threadIdx.x] = w2[threadIdx.x]; }
  if (threadIdx.x < HD) { sg[threadIdx.x] = ln_g[threadIdx.x]; sb[threadIdx.x] = ln_b[threadIdx.x]; }
  __syncthreads();
  int i = blockIdx.x * blockDim.x + threadIdx.x;
  if (i >= NN) return;
  float v[HD], mu = 0.f;
#pragma unroll
  for (int j = 0; j < HD; ++j) { v[j] = x[(size_t)i * HD + j]; mu += v[j]; }
  mu *= (1.f / HD);
  float var = 0.f;
#pragma unroll
  for (int j = 0; j < HD; ++j) { float d = v[j] - mu; var += d * d; }
  var *= (1.f / HD);
  float rs = rsqrtf(var + LN_EPS);
#pragma unroll
  for (int j = 0; j < HD; ++j) v[j] = (v[j] - mu) * rs * sg[j] + sb[j];
  float z = b2[0];
  for (int j = 0; j < 32; ++j) {
    float s = sb1[j];
#pragma unroll
    for (int k = 0; k < HD; ++k) s += v[k] * sw1[k * 32 + j];
    z += fmaxf(s, 0.f) * sw2[j];
  }
  out[i] = 1.f / (1.f + expf(-z));
}

__global__ void k_global_head(const float* __restrict__ sums, const float* __restrict__ cnt,
                              const float* __restrict__ w1, const float* __restrict__ b1,
                              const float* __restrict__ w2, const float* __restrict__ b2,
                              float* __restrict__ out) {
  int g = threadIdx.x;
  if (g >= NG) return;
  float inv = 1.f / fmaxf(cnt[g], 1.f);
  float v[HD];
#pragma unroll
  for (int j = 0; j < HD; ++j) v[j] = sums[g * HD + j] * inv;
  float hb[32];
  for (int j = 0; j < 32; ++j) {
    float s = b1[j];
#pragma unroll
    for (int k = 0; k < HD; ++k) s += v[k] * w1[k * 32 + j];
    hb[j] = fmaxf(s, 0.f);
  }
  for (int o = 0; o < 5; ++o) {
    float s = b2[o];
    for (int j = 0; j < 32; ++j) s += hb[j] * w2[j * 5 + o];
    out[g * 5 + o] = s;
  }
}

// ---------------- driver -----------------------------------------------------
extern "C" void kernel_launch(void* const* d_in, const int* in_sizes, int n_in,
                              void* d_out, int out_size, void* d_ws, size_t ws_size,
                              hipStream_t stream) {
  const float* x_in   = (const float*)d_in[0];
  const int*   eidx   = (const int*)  d_in[1];
  const float* eattr  = (const float*)d_in[2];
  const int*   batch  = (const int*)  d_in[3];
  const float* lnw    = (const float*)d_in[4];
  const float* lnb    = (const float*)d_in[5];
  const float* lew    = (const float*)d_in[6];
  const float* leb    = (const float*)d_in[7];
  const float* srcw   = (const float*)d_in[8];
  const float* srcb   = (const float*)d_in[9];
  const float* dstw   = (const float*)d_in[10];
  const float* dstb   = (const float*)d_in[11];
  const float* m1w    = (const float*)d_in[12];
  const float* m1b    = (const float*)d_in[13];
  const float* m2w    = (const float*)d_in[14];
  const float* m2b    = (const float*)d_in[15];
  const float* ln_g   = (const float*)d_in[16];
  const float* ln_b   = (const float*)d_in[17];
  const float* nhw1   = (const float*)d_in[18];
  const float* nhb1   = (const float*)d_in[19];
  const float* nhw2   = (const float*)d_in[20];
  const float* nhb2   = (const float*)d_in[21];
  const float* ghw1   = (const float*)d_in[22];
  const float* ghb1   = (const float*)d_in[23];
  const float* ghw2   = (const float*)d_in[24];
  const float* ghb2   = (const float*)d_in[25];
  float* out = (float*)d_out;

  char* w = (char*)d_ws;
  size_t off = 0;
  auto take = [&](size_t b) -> void* {
    void* p = w + off; off = (off + b + 511) & ~(size_t)511; return p;
  };
  float*    xF   = (float*)   take((size_t)NN * HD * 4);
  _Float16* x16  = (_Float16*)take((size_t)NN * HD * 2);
  _Float16* xs16 = (_Float16*)take((size_t)NN * HD * 2);
  _Float16* xd16 = (_Float16*)take((size_t)NN * HD * 2);
  float*    aggr = (float*)   take((size_t)NN * HD * 4);
  _Float16* ea16 = (_Float16*)take((size_t)NEL * HD * 2);
  int*      srcI = (int*)     take((size_t)NEL * 4);
  int*      dstI = (int*)     take((size_t)NEL * 4);
  float*    sums = (float*)   take((size_t)NG * HD * 4);
  float*    cnt  = (float*)   take((size_t)NG * 4);
  _Float16* wtN  = (_Float16*)take((size_t)HD * 64 * 2);   // lin_node: Kpad 64
  _Float16* wtE  = (_Float16*)take((size_t)HD * 96 * 2);   // lin_edge: Kpad 96
  _Float16* wtS  = (_Float16*)take((size_t)NL * HD * HD * 2);
  _Float16* wtD  = (_Float16*)take((size_t)NL * HD * HD * 2);
  _Float16* wtM1 = (_Float16*)take((size_t)NL * HD * 192 * 2);
  _Float16* wtM2 = (_Float16*)take((size_t)NL * HD * HD * 2);

  auto cdiv = [](int a, int b) { return (a + b - 1) / b; };

  // weight conversions (f32 -> f16, transposed, K zero-padded)
  k_conv_bt<<<cdiv(HD * 64, 256), 256, 0, stream>>>(lnw, wtN, KNODE, HD, 64);
  k_conv_bt<<<cdiv(HD * 96, 256), 256, 0, stream>>>(lew, wtE, KEDGE, HD, 96);
  for (int l = 0; l < NL; ++l) {
    k_conv_bt<<<cdiv(HD * HD, 256), 256, 0, stream>>>(srcw + (size_t)l * HD * HD, wtS + (size_t)l * HD * HD, HD, HD, HD);
    k_conv_bt<<<cdiv(HD * HD, 256), 256, 0, stream>>>(dstw + (size_t)l * HD * HD, wtD + (size_t)l * HD * HD, HD, HD, HD);
    k_conv_bt<<<cdiv(HD * 192, 256), 256, 0, stream>>>(m1w + (size_t)l * 192 * HD, wtM1 + (size_t)l * HD * 192, 192, HD, 192);
    k_conv_bt<<<cdiv(HD * HD, 256), 256, 0, stream>>>(m2w + (size_t)l * HD * HD, wtM2 + (size_t)l * HD * HD, HD, HD, HD);
  }

  k_build_edges<<<cdiv(NEL, 256), 256, 0, stream>>>(eidx, srcI, dstI);
  k_zero_ea_tail<<<cdiv(NN * HD, 256), 256, 0, stream>>>(ea16);

  // input projections (WMMA, padded K)
  k_in_proj<<<cdiv(NN / 16, 8), 256, 0, stream>>>(x_in, wtN, lnb, xF, (_Float16*)nullptr, KNODE, 64, NN);
  k_in_proj<<<cdiv(NE / 16, 8), 256, 0, stream>>>(eattr, wtE, leb, (float*)nullptr, ea16, KEDGE, 96, NE);

  for (int l = 0; l < NL; ++l) {
    k_f32_to_f16<<<cdiv(NN * HD, 256), 256, 0, stream>>>(xF, x16, NN * HD);
    k_node_lin<<<cdiv(NN / 16, 8), 256, 0, stream>>>(x16, wtS + (size_t)l * HD * HD, srcb + l * HD, xs16, NN);
    k_node_lin<<<cdiv(NN / 16, 8), 256, 0, stream>>>(x16, wtD + (size_t)l * HD * HD, dstb + l * HD, xd16, NN);
    hipMemsetAsync(aggr, 0, (size_t)NN * HD * 4, stream);
    k_edge_mlp<<<cdiv(NEL / 16, 8), 256, 0, stream>>>(xs16, xd16, ea16, srcI, dstI,
        wtM1 + (size_t)l * HD * 192, m1b + l * HD,
        wtM2 + (size_t)l * HD * HD,  m2b + l * HD, aggr, NEL);
    k_x_update<<<cdiv(NN * HD, 256), 256, 0, stream>>>(xF, aggr, NN * HD);
  }

  hipMemsetAsync(sums, 0, (size_t)NG * HD * 4, stream);
  hipMemsetAsync(cnt, 0, (size_t)NG * 4, stream);
  k_pool<<<cdiv(NN * HD, 256), 256, 0, stream>>>(xF, batch, sums, cnt);
  k_node_head<<<cdiv(NN, 256), 256, 0, stream>>>(xF, ln_g, ln_b, nhw1, nhb1, nhw2, nhb2, out);
  k_global_head<<<1, 64, 0, stream>>>(sums, cnt, ghw1, ghb1, ghw2, ghb2, out + NN);
}
